// GraphAttentionPooling_22737556865374
// MI455X (gfx1250) — compile-verified, matrix operand order checked by reading, not run on hardware
//
#include <hip/hip_runtime.h>
#include <hip/hip_bf16.h>

typedef __attribute__((ext_vector_type(16))) __bf16 v16bf;
typedef __attribute__((ext_vector_type(8)))  float  v8f;

#define HIDDEN 256
#define HH 128

static __device__ __forceinline__ unsigned short f2bf(float f) {
  // round-to-nearest-even f32 -> bf16
  unsigned u = __builtin_bit_cast(unsigned, f);
  return (unsigned short)((u + 0x7FFFu + ((u >> 16) & 1u)) >> 16);
}

// -------------------------------------------------------------------------
// Kernel 1: scores[i] = tanh(x[i,:] @ w1 + b1) @ w2 + b2   via bf16 WMMA
// Block = 256 threads = 8 waves; each wave owns a 16-row M-tile (128 rows/block)
// -------------------------------------------------------------------------
__global__ __launch_bounds__(256) void score_kernel(
    const float* __restrict__ x, const float* __restrict__ w1,
    const float* __restrict__ b1, const float* __restrict__ w2,
    const float* __restrict__ b2p, float* __restrict__ scores, int N)
{
  // w1 (256x128) pre-swizzled into per-lane bf16 B fragments: [kt][nt][lane][16]
  __shared__ unsigned short w1s[8 * 8 * 32 * 16]; // 64 KB

  const int tid = threadIdx.x;
  for (int e = tid; e < HIDDEN * HH; e += 256) {
    int k = e >> 7, n = e & 127;
    int kt = k >> 5, nt = n >> 4;
    int kk = k & 31, nn = n & 15;
    int hi = (kk >> 3) & 1;                       // K blocks 8..15 / 24..31 -> hi lanes
    int lane = nn + (hi << 4);
    int idx = (kk & 7) + ((kk & 16) ? 8 : 0);     // second 16-K half -> slots 8..15
    w1s[(((kt << 3) + nt) << 9) + (lane << 4) + idx] = f2bf(w1[e]);
  }
  __syncthreads();

  const int lane = tid & 31;
  const int wv   = tid >> 5;
  const int hi   = lane >> 4;
  const int mrow = lane & 15;
  const int rowbase = blockIdx.x * 128 + wv * 16;
  const int row = rowbase + mrow;
  const long long rowc = (row < N) ? row : (long long)(N - 1);
  const float* xr = x + rowc * (long long)HIDDEN;

  v8f acc[8] = {};

  #pragma unroll
  for (int kt = 0; kt < 8; ++kt) {
    const int ka = kt * 32 + hi * 8;   // first 8-K run for this lane half
    const int kb = ka + 16;            // second 8-K run
    float4 a0 = *reinterpret_cast<const float4*>(xr + ka);
    float4 a1 = *reinterpret_cast<const float4*>(xr + ka + 4);
    float4 a2 = *reinterpret_cast<const float4*>(xr + kb);
    float4 a3 = *reinterpret_cast<const float4*>(xr + kb + 4);
    union { v16bf v; unsigned short s[16]; } A;
    A.s[0]=f2bf(a0.x);  A.s[1]=f2bf(a0.y);  A.s[2]=f2bf(a0.z);  A.s[3]=f2bf(a0.w);
    A.s[4]=f2bf(a1.x);  A.s[5]=f2bf(a1.y);  A.s[6]=f2bf(a1.z);  A.s[7]=f2bf(a1.w);
    A.s[8]=f2bf(a2.x);  A.s[9]=f2bf(a2.y);  A.s[10]=f2bf(a2.z); A.s[11]=f2bf(a2.w);
    A.s[12]=f2bf(a3.x); A.s[13]=f2bf(a3.y); A.s[14]=f2bf(a3.z); A.s[15]=f2bf(a3.w);

    #pragma unroll
    for (int nt = 0; nt < 8; ++nt) {
      union { v16bf v; uint4 q[2]; } B;
      const uint4* bp = reinterpret_cast<const uint4*>(
          &w1s[(((kt << 3) + nt) << 9) + (lane << 4)]);
      B.q[0] = bp[0];
      B.q[1] = bp[1];
      acc[nt] = __builtin_amdgcn_wmma_f32_16x16x32_bf16(
          false, A.v, false, B.v, (short)0, acc[nt], false, false);
    }
  }

  // Epilogue: C layout -> lane holds column n = nt*16 + (lane&15), rows r + 8*hi.
  // score_row = sum_n tanh(h[n]) * w2[n]  (+ b2)
  const float b2 = b2p[0];
  float srow[8];
  #pragma unroll
  for (int r = 0; r < 8; ++r) srow[r] = 0.0f;
  #pragma unroll
  for (int nt = 0; nt < 8; ++nt) {
    const int n = nt * 16 + mrow;
    const float bb = b1[n];
    const float ww = w2[n];
    #pragma unroll
    for (int r = 0; r < 8; ++r)
      srow[r] += tanhf(acc[nt][r] + bb) * ww;
  }
  // reduce across the 16 lanes of each half (columns)
  #pragma unroll
  for (int r = 0; r < 8; ++r) {
    #pragma unroll
    for (int off = 1; off < 16; off <<= 1)
      srow[r] += __shfl_xor(srow[r], off, 32);
  }
  if (mrow == 0) {
    #pragma unroll
    for (int r = 0; r < 8; ++r) {
      const int orow = rowbase + r + hi * 8;
      if (orow < N) scores[orow] = srow[r] + b2;
    }
  }
}

// -------------------------------------------------------------------------
// Kernel 2: per-graph softmax stats (batch is sorted -> binary search bounds)
// -------------------------------------------------------------------------
__global__ __launch_bounds__(256) void stats_kernel(
    const int* __restrict__ batch, const float* __restrict__ scores,
    int* __restrict__ segLo, int* __restrict__ segHi,
    float* __restrict__ segMax, float* __restrict__ segInv, int N)
{
  const int g = blockIdx.x;
  const int t = threadIdx.x;

  int lo = 0, h = N;
  while (lo < h) { int mid = (lo + h) >> 1; if (batch[mid] < g) lo = mid + 1; else h = mid; }
  int lo2 = lo, h2 = N;
  while (lo2 < h2) { int mid = (lo2 + h2) >> 1; if (batch[mid] < g + 1) lo2 = mid + 1; else h2 = mid; }
  const int s0 = lo, s1 = lo2;

  __shared__ float red[256];
  float m = -3.402823466e38f;
  for (int i = s0 + t; i < s1; i += 256) m = fmaxf(m, scores[i]);
  red[t] = m; __syncthreads();
  for (int w = 128; w > 0; w >>= 1) { if (t < w) red[t] = fmaxf(red[t], red[t + w]); __syncthreads(); }
  const float mx = red[0];
  __syncthreads();

  float s = 0.0f;
  for (int i = s0 + t; i < s1; i += 256) s += expf(scores[i] - mx);
  red[t] = s; __syncthreads();
  for (int w = 128; w > 0; w >>= 1) { if (t < w) red[t] += red[t + w]; __syncthreads(); }

  if (t == 0) {
    segLo[g] = s0; segHi[g] = s1; segMax[g] = mx;
    segInv[g] = (s1 > s0) ? (1.0f / red[0]) : 0.0f;
  }
}

// -------------------------------------------------------------------------
// Kernel 3: out[g,:] = sum_i attn_i * x[i,:]  (thread t owns column t; coalesced)
// -------------------------------------------------------------------------
__global__ __launch_bounds__(256) void pool_kernel(
    const float* __restrict__ x, const float* __restrict__ scores,
    const int* __restrict__ segLo, const int* __restrict__ segHi,
    const float* __restrict__ segMax, const float* __restrict__ segInv,
    float* __restrict__ out)
{
  const int g = blockIdx.x;
  const int t = threadIdx.x;
  const int s0 = segLo[g], s1 = segHi[g];
  const float mx = segMax[g], inv = segInv[g];

  __shared__ float attn[256];
  float acc = 0.0f;
  for (int base = s0; base < s1; base += 256) {
    const int i = base + t;
    attn[t] = (i < s1) ? expf(scores[i] - mx) * inv : 0.0f;
    __syncthreads();
    const int cnt = min(256, s1 - base);
    #pragma unroll 4
    for (int j = 0; j < cnt; ++j)
      acc += x[(long long)(base + j) * HIDDEN + t] * attn[j];
    __syncthreads();
  }
  out[(long long)g * HIDDEN + t] = acc;
}

// -------------------------------------------------------------------------
extern "C" void kernel_launch(void* const* d_in, const int* in_sizes, int n_in,
                              void* d_out, int out_size, void* d_ws, size_t ws_size,
                              hipStream_t stream) {
  const float* x     = (const float*)d_in[0];
  const int*   batch = (const int*)d_in[1];
  const float* w1    = (const float*)d_in[2];
  const float* b1    = (const float*)d_in[3];
  const float* w2    = (const float*)d_in[4];
  const float* b2    = (const float*)d_in[5];
  float* out = (float*)d_out;

  const int N = in_sizes[1];            // 1,000,000 nodes
  const int B = out_size / HIDDEN;      // 1024 graphs

  float* scores = (float*)d_ws;         // N floats
  float* segMax = scores + N;           // B floats
  float* segInv = segMax + B;           // B floats
  int*   segLo  = (int*)(segInv + B);   // B ints
  int*   segHi  = segLo + B;            // B ints

  const int blocks1 = (N + 127) / 128;
  score_kernel<<<blocks1, 256, 0, stream>>>(x, w1, b1, w2, b2, scores, N);
  stats_kernel<<<B, 256, 0, stream>>>(batch, scores, segLo, segHi, segMax, segInv, N);
  pool_kernel <<<B, 256, 0, stream>>>(x, scores, segLo, segHi, segMax, segInv, out);
}
